// Attention_5042291605734
// MI455X (gfx1250) — matrix operand order repaired and checked by reading, not compiled, not run on hardware
//
#include <hip/hip_runtime.h>
#include <hip/hip_bf16.h>

// MI455X / gfx1250 attention: bf16 WMMA (16x16x32) everywhere, f32 accumulate.
// Async global->LDS staging (ASYNCcnt) with LDS double buffering.
// B=1, S=4096, DIM=INNER=1024, H=16, HD=64.

typedef __attribute__((ext_vector_type(16))) __bf16 v16bf;
typedef __attribute__((ext_vector_type(8)))  float  v8f;

#define S_LEN   4096
#define DIM_    1024
#define INNER_  1024
#define NH      16
#define HD_     64

union BF16x16 { uint4 u[2]; v16bf v; };

__device__ __forceinline__ unsigned short f32_to_bf16_bits(float f) {
    unsigned int u = __float_as_uint(f);
    unsigned int r = (u + 0x7FFFu + ((u >> 16) & 1u)) >> 16;  // RNE
    return (unsigned short)r;
}

// CDNA5 async copy: one b128 per lane, global -> LDS, tracked by ASYNCcnt.
// LDS offset = low 32 bits of the generic pointer to __shared__ memory.
__device__ __forceinline__ void async_b128_to_lds(const unsigned short* lds_dst,
                                                  const unsigned short* gsrc) {
    unsigned lds_off = (unsigned)(unsigned long long)lds_dst;
    unsigned long long ga = (unsigned long long)gsrc;
    asm volatile("global_load_async_to_lds_b128 %0, %1, off"
                 :: "v"(lds_off), "v"(ga) : "memory");
}
__device__ __forceinline__ void wait_async0() {
    asm volatile("s_wait_asynccnt 0x0" ::: "memory");
}
__device__ __forceinline__ void wave_lds_fence() {
    // Same-wave LDS ops complete in order; this stops compiler reordering
    // and drains DScnt so P-tile stores are visible to this wave's loads.
    asm volatile("s_wait_dscnt 0x0" ::: "memory");
}

// ---------------------------------------------------------------------------
// f32 -> bf16 bulk convert
// ---------------------------------------------------------------------------
__global__ void __launch_bounds__(256)
cvt_f32_bf16(const float* __restrict__ in, unsigned short* __restrict__ out, int n) {
    int i = blockIdx.x * blockDim.x + threadIdx.x;
    int stride = gridDim.x * blockDim.x;
    for (; i < n; i += stride) out[i] = f32_to_bf16_bits(in[i]);
}

// ---------------------------------------------------------------------------
// Tiled bf16 GEMM:  C[M,N] = A[M,K] * B[K,N] + bias[N]
// 256 threads (8 waves), tile 128(M) x 128(N), k-step 32, LDS double-buffered.
// Wave -> 32x64 subtile: 2 A-frags x 4 B-frags = 8 WMMAs/k-step, so each
// B-fragment load feeds two WMMAs. A tile staged via async b128 copies;
// B tile transposed on store so all fragment loads are contiguous b128 reads.
// ---------------------------------------------------------------------------
__global__ void __launch_bounds__(256)
gemm_bf16(const unsigned short* __restrict__ A,
          const unsigned short* __restrict__ B,
          const float* __restrict__ bias,
          void* __restrict__ Cout,
          int M, int K, int N, int out_is_bf16)
{
    __shared__ unsigned short ldsA[2][128 * 32];   // 2 x 8 KB
    __shared__ unsigned short ldsBT[2][128 * 32];  // 2 x 8 KB

    const int tid  = threadIdx.x;
    const int wave = tid >> 5;
    const int lane = tid & 31;
    const int half = lane >> 4;
    const int l16  = lane & 15;

    const int wgM  = blockIdx.x * 128;
    const int wgN  = blockIdx.y * 128;
    const int rloc = (wave >> 1) * 32;    // wave's 32-row band
    const int nloc = (wave & 1) * 64;     // wave's 64-col band

    // staging coordinates
    const int krow = tid >> 3, bch = tid & 7;   // B: 32 k-rows x 8 chunks of 16

    v8f acc[2][4];
#pragma unroll
    for (int i = 0; i < 2; ++i)
#pragma unroll
        for (int t = 0; t < 4; ++t)
#pragma unroll
            for (int e = 0; e < 8; ++e) acc[i][t][e] = 0.0f;

    const int nk = K / 32;

    // --- staging helper (buf, k0) ---
    auto stage = [&](int buf, int k0) {
        // A (128x32) row-major: two async b128 copies per thread
#pragma unroll
        for (int i = 0; i < 2; ++i) {
            int idx = tid + i * 256;
            int row = idx >> 2, ch = idx & 3;
            async_b128_to_lds(&ldsA[buf][row * 32 + ch * 8],
                              A + (size_t)(wgM + row) * K + k0 + ch * 8);
        }
        // B (32x128) transposed on store -> BT[128][32]
        const unsigned short* src = B + (size_t)(k0 + krow) * N + wgN + bch * 16;
        uint4 p0 = *(const uint4*)(src);
        uint4 p1 = *(const uint4*)(src + 8);
        unsigned short tmp[16];
        *(uint4*)(tmp)     = p0;
        *(uint4*)(tmp + 8) = p1;
#pragma unroll
        for (int i = 0; i < 16; ++i)
            ldsBT[buf][(bch * 16 + i) * 32 + krow] = tmp[i];
    };

    stage(0, 0);  // prologue

    for (int kk = 0; kk < nk; ++kk) {
        wait_async0();
        __syncthreads();
        if (kk + 1 < nk) stage((kk + 1) & 1, (kk + 1) * 32);

        const unsigned short* aB = ldsA[kk & 1];
        const unsigned short* bB = ldsBT[kk & 1];

        // Two A fragments (rows rloc..+15, rloc+16..+31)
        BF16x16 af[2];
#pragma unroll
        for (int i = 0; i < 2; ++i) {
            int row = rloc + i * 16 + l16;
            af[i].u[0] = *(const uint4*)(&aB[row * 32 + 8 * half]);
            af[i].u[1] = *(const uint4*)(&aB[row * 32 + 16 + 8 * half]);
        }
        // Four B fragments, each reused by two WMMAs
        BF16x16 bfr[4];
#pragma unroll
        for (int t = 0; t < 4; ++t) {
            int n = nloc + t * 16 + l16;
            bfr[t].u[0] = *(const uint4*)(&bB[n * 32 + 8 * half]);
            bfr[t].u[1] = *(const uint4*)(&bB[n * 32 + 16 + 8 * half]);
        }
#pragma unroll
        for (int t = 0; t < 4; ++t) {
            acc[0][t] = __builtin_amdgcn_wmma_f32_16x16x32_bf16(
                false, af[0].v, false, bfr[t].v, (short)0, acc[0][t], false, false);
            acc[1][t] = __builtin_amdgcn_wmma_f32_16x16x32_bf16(
                false, af[1].v, false, bfr[t].v, (short)0, acc[1][t], false, false);
        }
    }

    // Epilogue: C layout -> vgpr r holds M=r+8*half, N=lane&15
#pragma unroll
    for (int t = 0; t < 4; ++t) {
        int col = wgN + nloc + t * 16 + l16;
        float bv = bias[col];
#pragma unroll
        for (int i = 0; i < 2; ++i) {
#pragma unroll
            for (int r = 0; r < 8; ++r) {
                int row = wgM + rloc + i * 16 + r + 8 * half;
                float v = acc[i][t][r] + bv;
                if (out_is_bf16)
                    ((unsigned short*)Cout)[(size_t)row * N + col] = f32_to_bf16_bits(v);
                else
                    ((float*)Cout)[(size_t)row * N + col] = v;
            }
        }
    }
}

// ---------------------------------------------------------------------------
// Flash attention: grid = (S/128, H), 256 threads = 8 waves.
// Each wave: 16 queries x HD=64 output, streaming 32-key blocks with
// double-buffered async K staging. One workgroup barrier per key block.
// ---------------------------------------------------------------------------
__global__ void __launch_bounds__(256)
attn_fa(const unsigned short* __restrict__ Q,
        const unsigned short* __restrict__ Km,
        const unsigned short* __restrict__ Vm,
        unsigned short* __restrict__ Ob)
{
    __shared__ unsigned short ldsK[2][32 * 64];    // K block row-major
    __shared__ unsigned short ldsVT[2][64 * 32];   // V block transposed
    __shared__ unsigned short ldsP[8][16 * 32];    // per-wave P tiles

    const int tid  = threadIdx.x;
    const int wave = tid >> 5;
    const int lane = tid & 31;
    const int half = lane >> 4;
    const int l16  = lane & 15;

    const int h  = blockIdx.y;
    const int q0 = blockIdx.x * 128 + wave * 16;
    const size_t hoff = (size_t)h * HD_;

    const int srow = tid >> 3, sch = tid & 7;  // staging: 32 rows x 8 chunks of 8

    // Q fragments (two 16x32 A tiles covering d=0..63), loaded once
    BF16x16 qf[2];
    {
        const unsigned short* qrow = Q + (size_t)(q0 + l16) * INNER_ + hoff;
#pragma unroll
        for (int f = 0; f < 2; ++f) {
            qf[f].u[0] = *(const uint4*)(qrow + f * 32 + 8 * half);
            qf[f].u[1] = *(const uint4*)(qrow + f * 32 + 16 + 8 * half);
        }
    }

    float m[8], l[8];
    v8f   o[4];
#pragma unroll
    for (int r = 0; r < 8; ++r) { m[r] = -3.0e38f; l[r] = 0.0f; }
#pragma unroll
    for (int t = 0; t < 4; ++t)
#pragma unroll
        for (int e = 0; e < 8; ++e) o[t][e] = 0.0f;

    auto stage = [&](int buf, int kb) {
        // K (32x64) row-major: async b128 copy
        async_b128_to_lds(&ldsK[buf][srow * 64 + sch * 8],
                          Km + (size_t)(kb + srow) * INNER_ + hoff + sch * 8);
        // V transposed on store -> VT[64][32]
        const unsigned short* vs = Vm + (size_t)(kb + srow) * INNER_ + hoff + sch * 8;
        uint4 vv = *(const uint4*)vs;
        unsigned short tmp[8];
        *(uint4*)tmp = vv;
#pragma unroll
        for (int i = 0; i < 8; ++i)
            ldsVT[buf][(sch * 8 + i) * 32 + srow] = tmp[i];
    };

    const int nit = S_LEN / 32;
    stage(0, 0);  // prologue

    for (int it = 0; it < nit; ++it) {
        wait_async0();
        __syncthreads();
        if (it + 1 < nit) stage((it + 1) & 1, (it + 1) * 32);

        const unsigned short* Kb = ldsK[it & 1];
        const unsigned short* Vt = ldsVT[it & 1];

        // --- scores: two 16x16 key tiles, contracting over d=64 ---
        v8f sc[2];
#pragma unroll
        for (int kt = 0; kt < 2; ++kt) {
#pragma unroll
            for (int e = 0; e < 8; ++e) sc[kt][e] = 0.0f;
#pragma unroll
            for (int f = 0; f < 2; ++f) {
                BF16x16 kf;  // elem j -> K(d)=f*32+8*half+j (j<8), f*32+16+8*half+(j-8) (j>=8), N(key)=kt*16+l16
                int kr = kt * 16 + l16;
                kf.u[0] = *(const uint4*)(&Kb[kr * 64 + f * 32 + 8 * half]);
                kf.u[1] = *(const uint4*)(&Kb[kr * 64 + f * 32 + 16 + 8 * half]);
                sc[kt] = __builtin_amdgcn_wmma_f32_16x16x32_bf16(
                    false, qf[f].v, false, kf.v, (short)0, sc[kt], false, false);
            }
        }

        // --- online softmax (C layout: vgpr r holds row r+8*half) ---
        float alpha[8];
#pragma unroll
        for (int r = 0; r < 8; ++r) {
            float bm = fmaxf(sc[0][r], sc[1][r]);
            bm = fmaxf(bm, __shfl_xor(bm, 1, 32));
            bm = fmaxf(bm, __shfl_xor(bm, 2, 32));
            bm = fmaxf(bm, __shfl_xor(bm, 4, 32));
            bm = fmaxf(bm, __shfl_xor(bm, 8, 32));
            float mn = fmaxf(m[r], bm);
            float a  = __expf(m[r] - mn);
            m[r] = mn;
            alpha[r] = a;
            float p0 = __expf(sc[0][r] - mn);
            float p1 = __expf(sc[1][r] - mn);
            sc[0][r] = p0;
            sc[1][r] = p1;
            float rs = p0 + p1;
            rs += __shfl_xor(rs, 1, 32);
            rs += __shfl_xor(rs, 2, 32);
            rs += __shfl_xor(rs, 4, 32);
            rs += __shfl_xor(rs, 8, 32);
            l[r] = l[r] * a + rs;
        }
#pragma unroll
        for (int t = 0; t < 4; ++t)
#pragma unroll
            for (int r = 0; r < 8; ++r) o[t][r] *= alpha[r];

        // --- P (C-layout f32) -> bf16 row-major 16x32 in wave-private LDS ---
        unsigned short* pw = ldsP[wave];
#pragma unroll
        for (int r = 0; r < 8; ++r) {
            pw[(r + 8 * half) * 32 + l16]      = f32_to_bf16_bits(sc[0][r]);
            pw[(r + 8 * half) * 32 + 16 + l16] = f32_to_bf16_bits(sc[1][r]);
        }
        wave_lds_fence();  // same-wave in-order LDS: no workgroup barrier needed

        // --- PV: A = P (16x32 keys), B = V tiles (32 keys x 16 d) ---
        BF16x16 pf;
        pf.u[0] = *(const uint4*)(&pw[l16 * 32 + 8 * half]);
        pf.u[1] = *(const uint4*)(&pw[l16 * 32 + 16 + 8 * half]);
        BF16x16 vfr[4];
#pragma unroll
        for (int t = 0; t < 4; ++t) {
            vfr[t].u[0] = *(const uint4*)(&Vt[(t * 16 + l16) * 32 + 8 * half]);
            vfr[t].u[1] = *(const uint4*)(&Vt[(t * 16 + l16) * 32 + 16 + 8 * half]);
        }
#pragma unroll
        for (int t = 0; t < 4; ++t)
            o[t] = __builtin_amdgcn_wmma_f32_16x16x32_bf16(
                false, pf.v, false, vfr[t].v, (short)0, o[t], false, false);
    }

    // --- epilogue: normalize and store bf16 [S, INNER] ---
#pragma unroll
    for (int t = 0; t < 4; ++t) {
#pragma unroll
        for (int r = 0; r < 8; ++r) {
            int row = q0 + r + 8 * half;
            int col = (int)hoff + t * 16 + l16;
            Ob[(size_t)row * INNER_ + col] = f32_to_bf16_bits(o[t][r] / l[r]);
        }
    }
}

// ---------------------------------------------------------------------------
// Host-side orchestration
// ---------------------------------------------------------------------------
extern "C" void kernel_launch(void* const* d_in, const int* in_sizes, int n_in,
                              void* d_out, int out_size, void* d_ws, size_t ws_size,
                              hipStream_t stream) {
    const float* x  = (const float*)d_in[0];
    const float* Wq = (const float*)d_in[1];
    const float* bq = (const float*)d_in[2];
    const float* Wk = (const float*)d_in[3];
    const float* bk = (const float*)d_in[4];
    const float* Wv = (const float*)d_in[5];
    const float* bv = (const float*)d_in[6];
    const float* Wo = (const float*)d_in[7];
    const float* bo = (const float*)d_in[8];
    float* out = (float*)d_out;

    const size_t NX = (size_t)S_LEN * DIM_;    // 4M elems
    const size_t NW = (size_t)DIM_ * INNER_;   // 1M elems

    unsigned short* ws = (unsigned short*)d_ws;
    unsigned short* xb  = ws;            // x   bf16
    unsigned short* wqb = xb  + NX;      // Wq  bf16
    unsigned short* wkb = wqb + NW;
    unsigned short* wvb = wkb + NW;
    unsigned short* wob = wvb + NW;
    unsigned short* Qb  = wob + NW;      // Q   bf16 [S, INNER]
    unsigned short* Kb  = Qb  + NX;
    unsigned short* Vb  = Kb  + NX;
    unsigned short* Ab  = Vb  + NX;      // attn output bf16

    // f32 -> bf16 conversions
    cvt_f32_bf16<<<4096, 256, 0, stream>>>(x,  xb,  (int)NX);
    cvt_f32_bf16<<<2048, 256, 0, stream>>>(Wq, wqb, (int)NW);
    cvt_f32_bf16<<<2048, 256, 0, stream>>>(Wk, wkb, (int)NW);
    cvt_f32_bf16<<<2048, 256, 0, stream>>>(Wv, wvb, (int)NW);
    cvt_f32_bf16<<<2048, 256, 0, stream>>>(Wo, wob, (int)NW);

    // QKV projections (bf16 out for WMMA attention)
    dim3 ggrid(S_LEN / 128, INNER_ / 128);
    gemm_bf16<<<ggrid, 256, 0, stream>>>(xb, wqb, bq, Qb, S_LEN, DIM_, INNER_, 1);
    gemm_bf16<<<ggrid, 256, 0, stream>>>(xb, wkb, bk, Kb, S_LEN, DIM_, INNER_, 1);
    gemm_bf16<<<ggrid, 256, 0, stream>>>(xb, wvb, bv, Vb, S_LEN, DIM_, INNER_, 1);

    // Flash attention
    attn_fa<<<dim3(S_LEN / 128, NH), 256, 0, stream>>>(Qb, Kb, Vb, Ab);

    // Output projection (f32 out)
    gemm_bf16<<<dim3(S_LEN / 128, DIM_ / 128), 256, 0, stream>>>(
        Ab, wob, bo, out, S_LEN, INNER_, DIM_, 0);
}
